// BiLSTMTagger_49400713838819
// MI455X (gfx1250) — matrix operand order repaired and checked by reading, not compiled, not run on hardware
//
#include <hip/hip_runtime.h>
#include <hip/hip_bf16.h>

// ---------------------------------------------------------------------------
// BiLSTM tagger on gfx1250 (MI455X, wave32, WMMA).
// Sizes fixed by the reference: B=32, S=256, E=H=512, 4H=2048, V=50000, T=64.
// Strategy:
//   1) gather embeddings -> f16 X [S*B, E]   (row = s*32 + b)
//   2) pack all weights into 16x16x32 WMMA B-fragment layout (f16)
//   3) big WMMA GEMM:  XG[dir] = X @ Wx[dir] + bias   ([8192 x 2048] f32)
//   4) persistent recurrence: 1 workgroup (32 waves) per direction,
//      h in LDS, c in registers, h@Wh via v_wmma_f32_16x16x32_f16 per step.
//      XG gate loads are issued BEFORE the K-loop so their L2 latency hides
//      behind the WMMA chain (LOADcnt decrements in order on gfx1250).
//   5) repeat 3+4 for layer 2 (input = h1 [8192 x 1024] f16)
//   6) WMMA tag GEMM [8192x1024]@[1024x64] with (s,b)->(b,s) row remap
// ---------------------------------------------------------------------------

typedef __attribute__((ext_vector_type(16))) _Float16 v16h;
typedef __attribute__((ext_vector_type(8)))  float    v8f;

union F16x16 { v16h v; uint4 u[2]; };

__device__ inline v16h ld_frag2(const _Float16* p0, const _Float16* p1) {
  F16x16 t;
  t.u[0] = *(const uint4*)p0;
  t.u[1] = *(const uint4*)p1;
  return t.v;
}

__device__ inline v8f wmma_f16(v16h a, v16h b, v8f c) {
  // D = A(16x32 f16) * B(32x16 f16) + C(16x16 f32)
  return __builtin_amdgcn_wmma_f32_16x16x32_f16(
      /*neg_a=*/false, a, /*neg_b=*/false, b,
      /*c_mod=*/(short)0, c, /*reuse_a=*/false, /*reuse_b=*/false);
}

__device__ inline float dsigmoid(float x) { return 1.0f / (1.0f + __expf(-x)); }

// ------------------------- constants ---------------------------------------
#define SB    8192      // S*B rows
#define EDIM  512
#define HDIM  512
#define G4H   2048      // 4*H
#define SEQ   256
#define BATCH 32
#define TDIM  64

// ------------------------- kernel 1: embedding gather ----------------------
__global__ void gather_emb(const int* __restrict__ words,
                           const float* __restrict__ emb,
                           _Float16* __restrict__ X) {
  int r = blockIdx.x;            // row = s*32 + b
  int s = r >> 5, b = r & 31;
  int wi = words[b * SEQ + s];   // words is [B, S]
  const float* src = emb + (size_t)wi * EDIM;
  _Float16* dst = X + (size_t)r * EDIM;
  for (int c = threadIdx.x; c < EDIM; c += blockDim.x)
    dst[c] = (_Float16)src[c];
}

// ------------------------- kernel 2: weight fragment packing ---------------
// Dest layout: [kt][nt][lane(32)][e(16)] halfs, tile = kt*NT + nt.
// B-fragment element e of lane L holds  K = kt*32 + (L<16 ? 0 : 8) + (e<8 ? e : 8+e),
//                                       N = nt*16 + (L%16).
__global__ void pack_w(const float* __restrict__ src, _Float16* __restrict__ dst,
                       int KT, int NT, int Dtot, int rowOff, int tagMode, int Ncols) {
  int tid  = blockIdx.x * blockDim.x + threadIdx.x;
  int tile = tid >> 5, lane = tid & 31;
  if (tile >= KT * NT) return;
  int kt = tile / NT, nt = tile % NT;
  int kBase = kt * 32 + ((lane & 16) ? 8 : 0);
  int n = nt * 16 + (lane & 15);
  _Float16* d = dst + ((size_t)tile * 32 + lane) * 16;
#pragma unroll
  for (int e = 0; e < 16; ++e) {
    int k = kBase + (e < 8 ? e : e + 8);
    float v;
    if (tagMode) {
      v = src[(size_t)k * Ncols + n];                       // Wtag [K, T]
    } else {
      int gate = n >> 9, h = n & 511, dd = rowOff + k;      // W [4, Dtot, 512]
      v = src[((size_t)gate * Dtot + dd) * 512 + h];
    }
    d[e] = (_Float16)v;
  }
}

// ------------------------- kernel 3: generic WMMA GEMM ---------------------
// C[dir] = A @ B[dir] + bias[dir], A row-major f16 [M,K], B pre-packed frags,
// C f32 [M,N].  Each wave computes a 16x64 strip (1 A frag reused x4 B frags).
__global__ void __launch_bounds__(256) wmma_gemm(
    const _Float16* __restrict__ A, int lda,
    const _Float16* __restrict__ Bp, size_t bpDirStride,
    const float* __restrict__ bias0, const float* __restrict__ bias1,
    float* __restrict__ C, size_t cDirStride,
    int M, int N, int K, int remap) {
  const int lane = threadIdx.x & 31;
  const int wave = threadIdx.x >> 5;
  const int dir  = blockIdx.y;
  const _Float16* B = Bp + (size_t)dir * bpDirStride;
  const float* bias = dir ? bias1 : bias0;
  float* Cd = C + (size_t)dir * cDirStride;

  const int NT = N >> 4;
  const int KT = K >> 5;
  const int stripsN   = N >> 6;            // 4 N-tiles per strip
  const int numStrips = (M >> 4) * stripsN;
  const int ln  = lane & 15;
  const int hi8 = (lane & 16) ? 8 : 0;

  for (int strip = blockIdx.x * 8 + wave; strip < numStrips;
       strip += gridDim.x * 8) {
    int mt = strip / stripsN;
    int ntBase = (strip % stripsN) << 2;
    int m0 = mt << 4;

    v8f acc[4] = {};
    const _Float16* Arow = A + (size_t)(m0 + ln) * lda + hi8;
    for (int kt = 0; kt < KT; ++kt) {
      v16h a = ld_frag2(Arow + kt * 32, Arow + kt * 32 + 16);
      const _Float16* bp =
          B + (((size_t)kt * NT + ntBase) * 32 + lane) * 16;
      __builtin_prefetch((const void*)(bp + (size_t)NT * 512), 0, 1);
#pragma unroll
      for (int j = 0; j < 4; ++j) {
        v16h b = ld_frag2(bp + j * 512, bp + j * 512 + 8);
        acc[j] = wmma_f16(a, b, acc[j]);
      }
    }
#pragma unroll
    for (int j = 0; j < 4; ++j) {
      int n0 = (ntBase + j) << 4;
      float bs = bias ? bias[n0 + ln] : 0.0f;
#pragma unroll
      for (int r = 0; r < 8; ++r) {
        int gr = m0 + r + hi8;                        // row in (s*32+b) space
        int orow = remap ? ((gr & 31) * SEQ + (gr >> 5)) : gr;
        Cd[(size_t)orow * N + n0 + ln] = acc[j][r] + bs;
      }
    }
  }
}

// ------------------------- kernel 4: persistent LSTM recurrence ------------
// One workgroup (1024 thr = 32 waves) per direction. Wave w owns h-tile w
// (cols w*16..w*16+15) for both batch tiles (m=0..15 and 16..31); lane owns
// 16 cell-state values in registers for the whole sequence.
__global__ void __launch_bounds__(1024) lstm_recur(
    const float* __restrict__ XG,     // [2][8192][2048] (x-proj + bias)
    const _Float16* __restrict__ Whp, // [2] packed frags, KT=16, NT=128
    _Float16* __restrict__ Hout) {    // [8192][1024]
  __shared__ _Float16 hbuf[BATCH * HDIM];   // 32 KB

  const int dir = blockIdx.x;
  const int reverse = dir;
  const int colOff = dir * HDIM;
  const float* xg = XG + (size_t)dir * SB * G4H;
  const _Float16* Wh = Whp + (size_t)dir * (16 * 128 * 32 * 16);

  const int lane = threadIdx.x & 31;
  const int w    = threadIdx.x >> 5;        // wave id == h-tile index (0..31)
  const int ln   = lane & 15;
  const int hi8  = (lane & 16) ? 8 : 0;
  const int acol = (lane & 16) ? 8 : 0;
  const int hcol = w * 16 + ln;

  for (int i = threadIdx.x; i < BATCH * HDIM; i += 1024)
    hbuf[i] = (_Float16)0.0f;
  float creg[2][8];
#pragma unroll
  for (int q = 0; q < 2; ++q)
#pragma unroll
    for (int r = 0; r < 8; ++r) creg[q][r] = 0.0f;
  __syncthreads();

  for (int t = 0; t < SEQ; ++t) {
    int s = reverse ? (SEQ - 1 - t) : t;

    // ---- early-issue the XG gate loads for this step; LOADcnt decrements
    //      in order, so the K-loop's own load waits guarantee completion and
    //      the L2 latency hides behind the 128-WMMA chain below.
    float xgl[2][4][8];
#pragma unroll
    for (int q = 0; q < 2; ++q)
#pragma unroll
      for (int g = 0; g < 4; ++g)
#pragma unroll
        for (int r = 0; r < 8; ++r)
          xgl[q][g][r] =
              xg[(size_t)(s * BATCH + q * 16 + r + hi8) * G4H + g * HDIM + hcol];

    v8f acc[2][4] = {};                      // [batch tile][gate]
    for (int kt = 0; kt < 16; ++kt) {
      int col = kt * 32 + acol;
      const _Float16* ha0 = &hbuf[(size_t)(ln)      * HDIM + col];
      const _Float16* ha1 = &hbuf[(size_t)(16 + ln) * HDIM + col];
      v16h a0 = ld_frag2(ha0, ha0 + 16);
      v16h a1 = ld_frag2(ha1, ha1 + 16);
#pragma unroll
      for (int g = 0; g < 4; ++g) {
        const _Float16* bp =
            Wh + (((size_t)kt * 128 + g * 32 + w) * 32 + lane) * 16;
        v16h b = ld_frag2(bp, bp + 8);
        acc[0][g] = wmma_f16(a0, b, acc[0][g]);
        acc[1][g] = wmma_f16(a1, b, acc[1][g]);
      }
    }
    __syncthreads();   // all hbuf reads complete before overwrite

#pragma unroll
    for (int q = 0; q < 2; ++q) {
#pragma unroll
      for (int r = 0; r < 8; ++r) {
        int batch = q * 16 + r + hi8;
        float fg = dsigmoid(acc[q][0][r] + xgl[q][0][r]);
        float ig = dsigmoid(acc[q][1][r] + xgl[q][1][r]);
        float cg = tanhf  (acc[q][2][r] + xgl[q][2][r]);
        float og = dsigmoid(acc[q][3][r] + xgl[q][3][r]);
        float c = fg * creg[q][r] + ig * cg;
        creg[q][r] = c;
        float h = og * tanhf(c);
        hbuf[(size_t)batch * HDIM + hcol] = (_Float16)h;
        Hout[(size_t)(s * BATCH + batch) * (2 * HDIM) + colOff + hcol] =
            (_Float16)h;
      }
    }
    __syncthreads();   // new h visible before next step's reads
  }
}

// ------------------------- host launch -------------------------------------
extern "C" void kernel_launch(void* const* d_in, const int* in_sizes, int n_in,
                              void* d_out, int out_size, void* d_ws, size_t ws_size,
                              hipStream_t stream) {
  const int*   words = (const int*)d_in[0];
  // d_in[1] = lengths (unused by the reference computation)
  const float* emb   = (const float*)d_in[2];
  const float* Wf1   = (const float*)d_in[3];
  const float* bf1   = (const float*)d_in[4];
  const float* Wb1   = (const float*)d_in[5];
  const float* bb1   = (const float*)d_in[6];
  const float* Wf2   = (const float*)d_in[7];
  const float* bf2   = (const float*)d_in[8];
  const float* Wb2   = (const float*)d_in[9];
  const float* bb2   = (const float*)d_in[10];
  const float* Wtag  = (const float*)d_in[11];
  const float* btag  = (const float*)d_in[12];
  float* out = (float*)d_out;

  // ---- workspace carve (~197 MB) ----
  char* p = (char*)d_ws;
  _Float16* Xemb = (_Float16*)p; p += (size_t)SB * EDIM * 2;          // 8 MB
  _Float16* W1x  = (_Float16*)p; p += 2ull * 512 * G4H * 2;           // 4 MB
  _Float16* W1h  = (_Float16*)p; p += 2ull * 512 * G4H * 2;           // 4 MB
  _Float16* W2x  = (_Float16*)p; p += 2ull * 1024 * G4H * 2;          // 8 MB
  _Float16* W2h  = (_Float16*)p; p += 2ull * 512 * G4H * 2;           // 4 MB
  _Float16* Wtg  = (_Float16*)p; p += 1024ull * TDIM * 2;             // 128 KB
  float*    XG   = (float*)p;    p += 2ull * SB * G4H * 4;            // 128 MB
  _Float16* h1   = (_Float16*)p; p += (size_t)SB * 1024 * 2;          // 16 MB
  _Float16* h2   = (_Float16*)p; p += (size_t)SB * 1024 * 2;          // 16 MB
  (void)ws_size; (void)n_in; (void)in_sizes; (void)out_size;

  const size_t w1Stride = 512ull * G4H;     // halfs per direction (K=512)
  const size_t w2Stride = 1024ull * G4H;    // halfs per direction (K=1024)

  // 1) embeddings -> f16
  gather_emb<<<SB, 256, 0, stream>>>(words, emb, Xemb);

  // 2) pack weights into B-fragment layout
  auto packGrid = [](int KT, int NT) { return (KT * NT * 32 + 255) / 256; };
  pack_w<<<packGrid(16, 128), 256, 0, stream>>>(Wf1, W1x,            16, 128, 1024,    0, 0, 0);
  pack_w<<<packGrid(16, 128), 256, 0, stream>>>(Wb1, W1x + w1Stride, 16, 128, 1024,    0, 0, 0);
  pack_w<<<packGrid(16, 128), 256, 0, stream>>>(Wf1, W1h,            16, 128, 1024,  512, 0, 0);
  pack_w<<<packGrid(16, 128), 256, 0, stream>>>(Wb1, W1h + w1Stride, 16, 128, 1024,  512, 0, 0);
  pack_w<<<packGrid(32, 128), 256, 0, stream>>>(Wf2, W2x,            32, 128, 1536,    0, 0, 0);
  pack_w<<<packGrid(32, 128), 256, 0, stream>>>(Wb2, W2x + w2Stride, 32, 128, 1536,    0, 0, 0);
  pack_w<<<packGrid(16, 128), 256, 0, stream>>>(Wf2, W2h,            16, 128, 1536, 1024, 0, 0);
  pack_w<<<packGrid(16, 128), 256, 0, stream>>>(Wb2, W2h + w1Stride, 16, 128, 1536, 1024, 0, 0);
  pack_w<<<packGrid(32, 4),   256, 0, stream>>>(Wtag, Wtg,           32,   4,    0,    0, 1, TDIM);

  // 3) layer-1 input projections: XG[dir] = X @ W1x[dir] + b1[dir]
  wmma_gemm<<<dim3(2048, 2), 256, 0, stream>>>(
      Xemb, EDIM, W1x, w1Stride, bf1, bb1,
      XG, (size_t)SB * G4H, SB, G4H, 512, 0);

  // 4) layer-1 recurrence (dir = blockIdx.x; bwd time-reversed, cols 512..1023)
  lstm_recur<<<2, 1024, 0, stream>>>(XG, W1h, h1);

  // 5) layer-2 input projections: XG[dir] = h1 @ W2x[dir] + b2[dir]
  wmma_gemm<<<dim3(2048, 2), 256, 0, stream>>>(
      h1, 1024, W2x, w2Stride, bf2, bb2,
      XG, (size_t)SB * G4H, SB, G4H, 1024, 0);

  // 6) layer-2 recurrence
  lstm_recur<<<2, 1024, 0, stream>>>(XG, W2h, h2);

  // 7) tag projection with (s*32+b) -> (b*256+s) output row remap
  wmma_gemm<<<dim3(64, 1), 256, 0, stream>>>(
      h2, 1024, Wtg, 0, btag, btag,
      out, 0, SB, TDIM, 1024, 1);
}
